// GNNPolicy_MILP_63007170232493
// MI455X (gfx1250) — matrix operand order repaired and checked by reading, not compiled, not run on hardware
//
#include <hip/hip_runtime.h>
#include <hip/hip_bf16.h>

#define N_NODES 100000
#define N_EDGES 50000
#define DEG     16
#define NNZT    (N_EDGES * DEG)
#define HID     128
#define EMBD    64
#define NFREQ   48
#define FEAT    (2 * NFREQ)   // 96
#define DBLK    256
#define MT      4             // 16-row tiles per wave (B-fragment reuse factor)
#define ROWS    (MT * 16)     // 64 rows per wave

typedef _Float16 h8  __attribute__((ext_vector_type(8)));
typedef _Float16 h16 __attribute__((ext_vector_type(16)));
typedef float    f8  __attribute__((ext_vector_type(8)));

union H16u { h16 v; h8 h[2]; };

__device__ __forceinline__ f8 wmma16(h16 a, h16 b, f8 c) {
  // D = A(16x32 f16) * B(32x16 f16) + C(16x16 f32)
  return __builtin_amdgcn_wmma_f32_16x16x32_f16(false, a, false, b, (short)0, c,
                                                false, false);
}

// ---------------------------------------------------------------------------
// Multi-tile wave GEMM: out[MTILES*16][N] = act( in[MTILES*16][K] @ W + bias ).
// W given as WT [N][K] row-major f16; each B fragment is loaded once per k-step
// and reused for MTILES A tiles (4x less weight traffic from L2).
// A layout (ISA 7.12.2): lanes 0-15 hold K 0-7 & 16-23, lanes 16-31 hold
// K 8-15 & 24-31, M = lane&15. B: lane n<16 holds col n K 0-7 & 16-23, lane n+16
// K 8-15 & 24-31. C/D f32: VGPR r -> row r (+8 upper half), col = lane&15.
// ---------------------------------------------------------------------------
template <int MTILES, bool RELU>
__device__ __forceinline__ void wave_gemm_lds(const _Float16* __restrict__ inL, int K,
                                              const _Float16* __restrict__ WT,
                                              const float* __restrict__ bias,
                                              _Float16* __restrict__ outL, int N,
                                              int lane) {
  const int hs  = (lane >> 4) & 1;
  const int l15 = lane & 15;
  for (int n0 = 0; n0 < N; n0 += 16) {
    f8 acc[MTILES] = {};
    const _Float16* brow = WT + (size_t)(n0 + l15) * K + (hs ? 8 : 0);
    for (int k0 = 0; k0 < K; k0 += 32) {
      H16u b;
      b.h[0] = *(const h8*)(brow + k0);
      b.h[1] = *(const h8*)(brow + k0 + 16);
#pragma unroll
      for (int t = 0; t < MTILES; ++t) {
        const _Float16* arow = inL + (t * 16 + l15) * K + (hs ? 8 : 0) + k0;
        H16u a;
        a.h[0] = *(const h8*)(arow);
        a.h[1] = *(const h8*)(arow + 16);
        acc[t] = wmma16(a.v, b.v, acc[t]);
      }
    }
    float bb = bias[n0 + l15];
#pragma unroll
    for (int t = 0; t < MTILES; ++t) {
#pragma unroll
      for (int r = 0; r < 8; ++r) {
        float v = acc[t][r] + bb;
        if (RELU) v = fmaxf(v, 0.0f);
        outL[(t * 16 + r + hs * 8) * N + n0 + l15] = (_Float16)v;
      }
    }
  }
}

// Same, epilogue to global (f16/f32) with row-validity guard (nv = valid rows)
// and optional s_v-weighted column-sum accumulation (x_const.mean collapse).
template <int MTILES, bool RELU, bool MEAN, typename OT>
__device__ __forceinline__ void wave_gemm_out(const _Float16* __restrict__ inL, int K,
                                              const _Float16* __restrict__ WT,
                                              const float* __restrict__ bias,
                                              OT* __restrict__ outG, int ldo, int N,
                                              const float* __restrict__ wrowL,
                                              float* __restrict__ meanAcc,
                                              int lane, int nv) {
  const int hs  = (lane >> 4) & 1;
  const int l15 = lane & 15;
  for (int n0 = 0; n0 < N; n0 += 16) {
    f8 acc[MTILES] = {};
    const _Float16* brow = WT + (size_t)(n0 + l15) * K + (hs ? 8 : 0);
    for (int k0 = 0; k0 < K; k0 += 32) {
      H16u b;
      b.h[0] = *(const h8*)(brow + k0);
      b.h[1] = *(const h8*)(brow + k0 + 16);
#pragma unroll
      for (int t = 0; t < MTILES; ++t) {
        const _Float16* arow = inL + (t * 16 + l15) * K + (hs ? 8 : 0) + k0;
        H16u a;
        a.h[0] = *(const h8*)(arow);
        a.h[1] = *(const h8*)(arow + 16);
        acc[t] = wmma16(a.v, b.v, acc[t]);
      }
    }
    float bb   = bias[n0 + l15];
    float part = 0.0f;
#pragma unroll
    for (int t = 0; t < MTILES; ++t) {
#pragma unroll
      for (int r = 0; r < 8; ++r) {
        float v = acc[t][r] + bb;
        if (RELU) v = fmaxf(v, 0.0f);
        int row = t * 16 + r + hs * 8;
        if (row < nv) {
          outG[(size_t)row * ldo + n0 + l15] = (OT)v;
          if (MEAN) part += wrowL[row] * v;
        }
      }
    }
    if (MEAN) atomicAdd(&meanAcc[n0 + l15], part);
  }
}

// PeriodicEmbeddings (cos|sin) -> NLinear(96->64) -> 64->256 -> 256->256 (relu).
// Result [ROWS][256] left in bufB; caller applies the final 256->128 layer.
template <int MTILES>
__device__ __forceinline__ void emb_body(const float* __restrict__ zL,
                                         const float* __restrict__ pc,
                                         const _Float16* __restrict__ nwT,
                                         const float* __restrict__ nb,
                                         const _Float16* __restrict__ mw1T,
                                         const float* __restrict__ mb1,
                                         const _Float16* __restrict__ mw2T,
                                         const float* __restrict__ mb2,
                                         _Float16* bufA, _Float16* bufB, int lane) {
  for (int idx = lane; idx < MTILES * 16 * FEAT; idx += 32) {
    int row = idx / FEAT;
    int f   = idx - row * FEAT;
    float p = 6.283185307179586f * zL[row] * pc[f < NFREQ ? f : f - NFREQ];
    bufA[row * FEAT + f] = (_Float16)((f < NFREQ) ? __cosf(p) : __sinf(p));
  }
  wave_gemm_lds<MTILES, true>(bufA, FEAT, nwT, nb, bufB, EMBD, lane);   // ->64
  wave_gemm_lds<MTILES, true>(bufB, EMBD, mw1T, mb1, bufA, DBLK, lane); // ->256
  wave_gemm_lds<MTILES, true>(bufA, DBLK, mw2T, mb2, bufB, DBLK, lane); // ->256
}

// ---------------------------------------------------------------------------
// Graph-scalar kernels
// ---------------------------------------------------------------------------
__global__ void k_constraints(const int* __restrict__ row, const float* __restrict__ coef,
                              const float* __restrict__ rhs0, float* __restrict__ scaled,
                              float* __restrict__ rhs1, float* __restrict__ s_v,
                              float* __restrict__ sig) {
  __shared__ float ls[2];
  if (threadIdx.x < 2) ls[threadIdx.x] = 0.0f;
  __syncthreads();
  int e = blockIdx.x * blockDim.x + threadIdx.x;
  float r1 = 0.0f, r2 = 0.0f;
  if (e < N_EDGES) {
    float s = 0.0f;
#pragma unroll
    for (int j = 0; j < DEG; ++j) s += fabsf(coef[e * DEG + j]);
    float inv = (s == 0.0f) ? 0.0f : 1.0f / s;
#pragma unroll
    for (int j = 0; j < DEG; ++j) {
      float sc = coef[e * DEG + j] * inv;
      scaled[e * DEG + j] = sc;
      atomicAdd(&s_v[row[e * DEG + j]], sc);
    }
    float r = rhs0[e] * inv;
    rhs1[e] = r;
    r1 = r;
    r2 = r * inv;
  }
  atomicAdd(&ls[0], r1);
  atomicAdd(&ls[1], r2);
  __syncthreads();
  if (threadIdx.x == 0) { atomicAdd(&sig[0], ls[0]); atomicAdd(&sig[1], ls[1]); }
}

__global__ void k_inv_sv(const float* __restrict__ s_v, float* __restrict__ inv_s_v) {
  int v = blockIdx.x * blockDim.x + threadIdx.x;
  if (v >= N_NODES) return;
  float s = s_v[v];
  inv_s_v[v] = (s == 0.0f) ? 0.0f : 1.0f / s;
}

__global__ void k_x0(const int* __restrict__ row, const float* __restrict__ scaled,
                     const float* __restrict__ rhs1, const float* __restrict__ inv_s_v,
                     float* __restrict__ x0) {
  int i = blockIdx.x * blockDim.x + threadIdx.x;
  if (i >= NNZT) return;
  int r = row[i];
  atomicAdd(&x0[r], inv_s_v[r] * scaled[i] * rhs1[i >> 4]);  // col = i/16
}

__global__ void k_transpose_f16(const float* __restrict__ src, _Float16* __restrict__ dst,
                                int K, int N) {
  int idx = blockIdx.x * blockDim.x + threadIdx.x;
  if (idx >= K * N) return;
  int k = idx / N, n = idx - k * N;
  dst[(size_t)n * K + k] = (_Float16)src[idx];
}

// ---------------------------------------------------------------------------
// Big per-node embedding: X = emb(x0) (+ mean1 += s_v-weighted colsum),
// EV = emb(inv_s_v). One wave per 64-node tile, 2 waves per block.
// ---------------------------------------------------------------------------
__global__ __launch_bounds__(64) void k_emb_big(
    const float* __restrict__ x0, const float* __restrict__ inv_s_v,
    const float* __restrict__ s_v, const float* __restrict__ pc,
    const _Float16* __restrict__ nwT, const float* __restrict__ nb,
    const _Float16* __restrict__ mw1T, const float* __restrict__ mb1,
    const _Float16* __restrict__ mw2T, const float* __restrict__ mb2,
    const _Float16* __restrict__ mw3T, const float* __restrict__ mb3,
    _Float16* __restrict__ X, _Float16* __restrict__ EV, float* __restrict__ mean1) {
  __shared__ _Float16 bufA[2][ROWS * DBLK];  // 32KB per wave
  __shared__ _Float16 bufB[2][ROWS * DBLK];  // 32KB per wave
  __shared__ float    zL[2][ROWS];
  __shared__ float    wL[2][ROWS];
  int wid   = threadIdx.x >> 5;
  int lane  = threadIdx.x & 31;
  int tile  = blockIdx.x * 2 + wid;
  int node0 = tile * ROWS;
  if (node0 >= N_NODES) return;
  int nv = N_NODES - node0;
  if (nv > ROWS) nv = ROWS;
  for (int i = lane; i < ROWS; i += 32) {
    int n  = node0 + i;
    int nc = (n < N_NODES) ? n : (N_NODES - 1);
    zL[wid][i] = x0[nc];
    wL[wid][i] = (i < nv) ? s_v[nc] : 0.0f;
  }
  emb_body<MT>(zL[wid], pc, nwT, nb, mw1T, mb1, mw2T, mb2, bufA[wid], bufB[wid], lane);
  wave_gemm_out<MT, false, true, _Float16>(bufB[wid], DBLK, mw3T, mb3,
                                           X + (size_t)node0 * HID, HID, HID,
                                           wL[wid], mean1, lane, nv);
  for (int i = lane; i < ROWS; i += 32) {
    int n = node0 + i;
    zL[wid][i] = inv_s_v[(n < N_NODES) ? n : (N_NODES - 1)];
  }
  emb_body<MT>(zL[wid], pc, nwT, nb, mw1T, mb1, mw2T, mb2, bufA[wid], bufB[wid], lane);
  wave_gemm_out<MT, false, false, _Float16>(bufB[wid], DBLK, mw3T, mb3,
                                            EV + (size_t)node0 * HID, HID, HID,
                                            nullptr, nullptr, lane, nv);
}

// emb() of the two rhs-sum scalars: rows {sig1, sig2, 0,...}; f32 rows 0,1 -> sr.
__global__ void k_emb_scalar(const float* __restrict__ sig, const float* __restrict__ pc,
                             const _Float16* __restrict__ nwT, const float* __restrict__ nb,
                             const _Float16* __restrict__ mw1T, const float* __restrict__ mb1,
                             const _Float16* __restrict__ mw2T, const float* __restrict__ mb2,
                             const _Float16* __restrict__ mw3T, const float* __restrict__ mb3,
                             float* __restrict__ sr /* [2*HID] */) {
  __shared__ _Float16 bufA[16 * DBLK];
  __shared__ _Float16 bufB[16 * DBLK];
  __shared__ float    zL[16];
  __shared__ float    out32[16 * HID];
  int lane = threadIdx.x;
  if (lane < 16) zL[lane] = (lane < 2) ? sig[lane] : 0.0f;
  emb_body<1>(zL, pc, nwT, nb, mw1T, mb1, mw2T, mb2, bufA, bufB, lane);
  wave_gemm_out<1, false, false, float>(bufB, DBLK, mw3T, mb3, out32, HID, HID,
                                        nullptr, nullptr, lane, 16);
  for (int i = lane; i < 2 * HID; i += 32) sr[i] = out32[i];
}

// b[c] = sr[c] - ( (1/E) * sum_k mean[k]*lcw[k][c] + lcb[c] )
__global__ void k_bvec(const float* __restrict__ sr, const float* __restrict__ mean,
                       const float* __restrict__ lcw, const float* __restrict__ lcb,
                       float* __restrict__ b) {
  int c = threadIdx.x;
  float acc = 0.0f;
  for (int k = 0; k < HID; ++k) acc += mean[k] * lcw[k * HID + c];
  b[c] = sr[c] - (acc * (1.0f / N_EDGES) + lcb[c]);
}

// Conv-1 tail: X <- relu( (b1*EV + X) @ act_w0 + act_b0 ); mean2 += weighted.
__global__ __launch_bounds__(64) void k_update(
    _Float16* __restrict__ X, const _Float16* __restrict__ EV,
    const float* __restrict__ s_v, const float* __restrict__ b1,
    const _Float16* __restrict__ awT, const float* __restrict__ ab,
    float* __restrict__ mean2) {
  __shared__ _Float16 bufA[2][ROWS * HID];  // 16KB per wave
  __shared__ float    wL[2][ROWS];
  int wid   = threadIdx.x >> 5;
  int lane  = threadIdx.x & 31;
  int tile  = blockIdx.x * 2 + wid;
  int node0 = tile * ROWS;
  if (node0 >= N_NODES) return;
  int nv = N_NODES - node0;
  if (nv > ROWS) nv = ROWS;
  for (int i = lane; i < ROWS; i += 32)
    wL[wid][i] = (node0 + i < N_NODES) ? s_v[node0 + i] : 0.0f;
  for (int idx = lane; idx < ROWS * HID; idx += 32) {
    int rn = node0 + (idx >> 7);
    if (rn >= N_NODES) rn = N_NODES - 1;  // clamp loads inside workspace
    int    c = idx & (HID - 1);
    size_t g = (size_t)rn * HID + c;
    bufA[wid][idx] = (_Float16)(b1[c] * (float)EV[g] + (float)X[g]);
  }
  wave_gemm_out<MT, true, true, _Float16>(bufA[wid], HID, awT, ab,
                                          X + (size_t)node0 * HID, HID, HID,
                                          wL[wid], mean2, lane, nv);
}

// Conv-2 tail + output head, fused per 64-node tile.
__global__ __launch_bounds__(64) void k_head(
    const _Float16* __restrict__ X, const _Float16* __restrict__ EV,
    const float* __restrict__ b2, const _Float16* __restrict__ aw1T,
    const float* __restrict__ ab1, const _Float16* __restrict__ ow1T,
    const float* __restrict__ ob1, const _Float16* __restrict__ ow2T,
    const float* __restrict__ ob2, const float* __restrict__ ow3,
    const float* __restrict__ ob3, float* __restrict__ out) {
  __shared__ _Float16 bufA[2][ROWS * HID];
  __shared__ _Float16 bufB[2][ROWS * HID];
  int wid   = threadIdx.x >> 5;
  int lane  = threadIdx.x & 31;
  int tile  = blockIdx.x * 2 + wid;
  int node0 = tile * ROWS;
  if (node0 >= N_NODES) return;
  for (int idx = lane; idx < ROWS * HID; idx += 32) {
    int rn = node0 + (idx >> 7);
    if (rn >= N_NODES) rn = N_NODES - 1;
    int    c = idx & (HID - 1);
    size_t g = (size_t)rn * HID + c;
    bufA[wid][idx] = (_Float16)(b2[c] * (float)EV[g] + (float)X[g]);
  }
  wave_gemm_lds<MT, true>(bufA[wid], HID, aw1T, ab1, bufB[wid], HID, lane); // conv-2
  wave_gemm_lds<MT, true>(bufB[wid], HID, ow1T, ob1, bufA[wid], HID, lane); // h1
  wave_gemm_lds<MT, true>(bufA[wid], HID, ow2T, ob2, bufB[wid], HID, lane); // h2
  for (int row = lane; row < ROWS; row += 32) {
    if (node0 + row < N_NODES) {
      float acc = ob3[0];
      for (int k = 0; k < HID; ++k) acc += (float)bufB[wid][row * HID + k] * ow3[k];
      out[node0 + row] = acc;
    }
  }
}

// ---------------------------------------------------------------------------
extern "C" void kernel_launch(void* const* d_in, const int* in_sizes, int n_in,
                              void* d_out, int out_size, void* d_ws, size_t ws_size,
                              hipStream_t stream) {
  const int*   row  = (const int*)d_in[0];  // hyperedge_index[0] = rows; col = nnz/16
  const float* coef = (const float*)d_in[1];
  const float* rhs0 = (const float*)d_in[2];
  const float* pc   = (const float*)d_in[3];
  const float* nw   = (const float*)d_in[4];
  const float* nb   = (const float*)d_in[5];
  const float* mw1  = (const float*)d_in[6];
  const float* mb1  = (const float*)d_in[7];
  const float* mw2  = (const float*)d_in[8];
  const float* mb2  = (const float*)d_in[9];
  const float* mw3  = (const float*)d_in[10];
  const float* mb3  = (const float*)d_in[11];
  const float* lcw  = (const float*)d_in[12];
  const float* lcb  = (const float*)d_in[13];
  const float* aw   = (const float*)d_in[14];
  const float* ab   = (const float*)d_in[15];
  const float* ow1  = (const float*)d_in[16];
  const float* ob1  = (const float*)d_in[17];
  const float* ow2  = (const float*)d_in[18];
  const float* ob2  = (const float*)d_in[19];
  const float* ow3  = (const float*)d_in[20];
  const float* ob3  = (const float*)d_in[21];

  uint8_t* base = (uint8_t*)d_ws;
  size_t   off  = 0;
  auto alloc = [&](size_t bytes) -> void* {
    void* p = base + off;
    off = (off + bytes + 255) & ~(size_t)255;
    return p;
  };
  float*    scaled  = (float*)alloc((size_t)NNZT * 4);
  float*    rhs1    = (float*)alloc((size_t)N_EDGES * 4);
  float*    s_v     = (float*)alloc((size_t)N_NODES * 4);
  float*    inv_s_v = (float*)alloc((size_t)N_NODES * 4);
  float*    x0      = (float*)alloc((size_t)N_NODES * 4);
  float*    sig     = (float*)alloc(2 * 4);
  float*    mean1   = (float*)alloc(HID * 4);
  float*    mean2   = (float*)alloc(HID * 4);
  float*    sr      = (float*)alloc(2 * HID * 4);
  float*    b1      = (float*)alloc(HID * 4);
  float*    b2      = (float*)alloc(HID * 4);
  _Float16* nwT     = (_Float16*)alloc((size_t)FEAT * EMBD * 2);
  _Float16* mw1T    = (_Float16*)alloc((size_t)EMBD * DBLK * 2);
  _Float16* mw2T    = (_Float16*)alloc((size_t)DBLK * DBLK * 2);
  _Float16* mw3T    = (_Float16*)alloc((size_t)DBLK * HID * 2);
  _Float16* aw0T    = (_Float16*)alloc((size_t)HID * HID * 2);
  _Float16* aw1T    = (_Float16*)alloc((size_t)HID * HID * 2);
  _Float16* ow1T    = (_Float16*)alloc((size_t)HID * HID * 2);
  _Float16* ow2T    = (_Float16*)alloc((size_t)HID * HID * 2);
  _Float16* X       = (_Float16*)alloc((size_t)N_NODES * HID * 2);
  _Float16* EV      = (_Float16*)alloc((size_t)N_NODES * HID * 2 + 256);
  if (off > ws_size) return;  // scratch too small; nothing safe to do

  auto tpose = [&](const float* src, _Float16* dst, int K, int N) {
    int n = K * N;
    k_transpose_f16<<<(n + 255) / 256, 256, 0, stream>>>(src, dst, K, N);
  };
  tpose(nw, nwT, FEAT, EMBD);
  tpose(mw1, mw1T, EMBD, DBLK);
  tpose(mw2, mw2T, DBLK, DBLK);
  tpose(mw3, mw3T, DBLK, HID);
  tpose(aw, aw0T, HID, HID);
  tpose(aw + HID * HID, aw1T, HID, HID);
  tpose(ow1, ow1T, HID, HID);
  tpose(ow2, ow2T, HID, HID);

  hipMemsetAsync(s_v, 0, (size_t)N_NODES * 4, stream);
  hipMemsetAsync(x0, 0, (size_t)N_NODES * 4, stream);
  hipMemsetAsync(sig, 0, 2 * 4, stream);
  hipMemsetAsync(mean1, 0, HID * 4, stream);
  hipMemsetAsync(mean2, 0, HID * 4, stream);

  k_constraints<<<(N_EDGES + 255) / 256, 256, 0, stream>>>(row, coef, rhs0, scaled,
                                                           rhs1, s_v, sig);
  k_inv_sv<<<(N_NODES + 255) / 256, 256, 0, stream>>>(s_v, inv_s_v);
  k_x0<<<(NNZT + 255) / 256, 256, 0, stream>>>(row, scaled, rhs1, inv_s_v, x0);

  k_emb_scalar<<<1, 32, 0, stream>>>(sig, pc, nwT, nb, mw1T, mb1, mw2T, mb2, mw3T, mb3, sr);

  const int nWaveTiles = (N_NODES + ROWS - 1) / ROWS;  // 1563
  const int nBlocks    = (nWaveTiles + 1) / 2;         // 782, 2 waves/block
  k_emb_big<<<nBlocks, 64, 0, stream>>>(x0, inv_s_v, s_v, pc, nwT, nb, mw1T, mb1,
                                        mw2T, mb2, mw3T, mb3, X, EV, mean1);

  k_bvec<<<1, HID, 0, stream>>>(sr, mean1, lcw, lcb, b1);
  k_update<<<nBlocks, 64, 0, stream>>>(X, EV, s_v, b1, aw0T, ab, mean2);
  k_bvec<<<1, HID, 0, stream>>>(sr + HID, mean2, lcw + HID * HID, lcb + HID, b2);
  k_head<<<nBlocks, 64, 0, stream>>>(X, EV, b2, aw1T, ab + HID, ow1T, ob1, ow2T, ob2,
                                     ow3, ob3, (float*)d_out);
}
